// BERT_22823456211847
// MI455X (gfx1250) — compile-verified, matrix operand order checked by reading, not run on hardware
//
#include <hip/hip_runtime.h>
#include <math.h>

// ---- problem constants (match reference) ----
#define B_      16
#define S_DIM   512
#define D_DIM   768
#define H_DIM   12
#define L_DIM   12
#define V_DIM   32000
#define P_DIM   20
#define NROWS   (B_ * S_DIM)   // 8192 token rows

typedef __attribute__((ext_vector_type(8)))  float  v8f;
typedef __attribute__((ext_vector_type(16))) __bf16 v16bf;

union Frag16 { v16bf v; uint4 u[2]; unsigned short us[16]; };

__device__ __forceinline__ unsigned short f2bf(float f) {
    union { float f; unsigned int u; } c; c.f = f;
    unsigned int r = c.u + 0x7FFFu + ((c.u >> 16) & 1u);   // RNE
    return (unsigned short)(r >> 16);
}
__device__ __forceinline__ float bf2f(unsigned short h) {
    union { unsigned int u; float f; } c; c.u = ((unsigned int)h) << 16;
    return c.f;
}

// ---- shared epilogue: one 16x16 C-tile element write ----
__device__ __forceinline__ void epilogue_store(
    int mode, int rowg, int colg, float val, int N,
    float* __restrict__ outF, unsigned short* __restrict__ outH,
    const float* __restrict__ resid)
{
    if (mode == 0) {
        outF[(size_t)rowg * N + colg] = val;
    } else if (mode == 1) {   // Q/K -> bf16 [B,H,S,64]
        int bb_ = rowg >> 9, s = rowg & 511, hh = colg >> 6, d = colg & 63;
        outH[((((size_t)bb_ * H_DIM + hh) * S_DIM + s) << 6) + d] = f2bf(val);
    } else if (mode == 2) {   // V^T -> bf16 [B,H,64,S]
        int bb_ = rowg >> 9, s = rowg & 511, hh = colg >> 6, d = colg & 63;
        outH[((((size_t)bb_ * H_DIM + hh) << 6) + d) * S_DIM + s] = f2bf(val);
    } else if (mode == 3) {   // pre-LN y = acc + bias + residual
        size_t idx = (size_t)rowg * N + colg;
        outF[idx] = val + resid[idx];
    } else if (mode == 5) {   // pooler
        outH[(size_t)rowg * N + colg] = f2bf(tanhf(val));
    } else {                  // 6: exact GELU
        outF[(size_t)rowg * N + colg] =
            val * 0.5f * (1.0f + erff(val * 0.70710678118654752f));
    }
}

// ---------------------------------------------------------------------------
// Main WMMA GEMM: D[M,N] = A[M,K] * Bt[N,K]^T (+bias, epilogue).
// One wave computes a 32x64 register tile: 2 A-frags x 4 B-frags -> 8 wmma
// per 32-wide K-chunk from 12 b128 loads (1.5 loads/wmma vs 4 for a 16x16
// tile) -> matrix-core bound instead of L2-bandwidth bound.
// Requires M % 32 == 0, N % 64 == 0, K % 32 == 0.
// ---------------------------------------------------------------------------
__global__ __launch_bounds__(256) void gemm32x64_wmma_bf16(
    const unsigned short* __restrict__ A, const unsigned short* __restrict__ Bt,
    const float* __restrict__ bias, float* __restrict__ outF,
    unsigned short* __restrict__ outH, const float* __restrict__ resid,
    int M, int N, int K, int mode)
{
    const int wave = threadIdx.x >> 5;
    const int lane = threadIdx.x & 31;
    const int half = lane >> 4;
    const int l    = lane & 15;
    const int tilesN = N >> 6;                 // 64-wide
    const int tiles  = (M >> 5) * tilesN;      // 32-tall
    const int id = blockIdx.x * 8 + wave;
    if (id >= tiles) return;
    const int tm = id / tilesN, tn = id - tm * tilesN;
    const int row0 = tm << 5, n0 = tn << 6;

    const unsigned short* arow0 = A  + (size_t)(row0 +  0 + l) * K + half * 8;
    const unsigned short* arow1 = A  + (size_t)(row0 + 16 + l) * K + half * 8;
    const unsigned short* brow0 = Bt + (size_t)(n0   +  0 + l) * K + half * 8;
    const unsigned short* brow1 = Bt + (size_t)(n0   + 16 + l) * K + half * 8;
    const unsigned short* brow2 = Bt + (size_t)(n0   + 32 + l) * K + half * 8;
    const unsigned short* brow3 = Bt + (size_t)(n0   + 48 + l) * K + half * 8;

    v8f acc[2][4] = {};
    for (int k0 = 0; k0 < K; k0 += 32) {
        Frag16 a0, a1, b0, b1, b2, b3;
        const uint4* p;
        p = (const uint4*)(arow0 + k0); a0.u[0] = p[0]; a0.u[1] = p[2];
        p = (const uint4*)(arow1 + k0); a1.u[0] = p[0]; a1.u[1] = p[2];
        p = (const uint4*)(brow0 + k0); b0.u[0] = p[0]; b0.u[1] = p[2];
        p = (const uint4*)(brow1 + k0); b1.u[0] = p[0]; b1.u[1] = p[2];
        p = (const uint4*)(brow2 + k0); b2.u[0] = p[0]; b2.u[1] = p[2];
        p = (const uint4*)(brow3 + k0); b3.u[0] = p[0]; b3.u[1] = p[2];
        __builtin_prefetch(arow0 + k0 + 64, 0, 3);   // global_prefetch_b8 (near)
        __builtin_prefetch(brow0 + k0 + 64, 0, 3);
        __builtin_prefetch(brow2 + k0 + 64, 0, 3);
        acc[0][0] = __builtin_amdgcn_wmma_f32_16x16x32_bf16(false, a0.v, false, b0.v, (short)0, acc[0][0], false, false);
        acc[0][1] = __builtin_amdgcn_wmma_f32_16x16x32_bf16(false, a0.v, false, b1.v, (short)0, acc[0][1], false, false);
        acc[0][2] = __builtin_amdgcn_wmma_f32_16x16x32_bf16(false, a0.v, false, b2.v, (short)0, acc[0][2], false, false);
        acc[0][3] = __builtin_amdgcn_wmma_f32_16x16x32_bf16(false, a0.v, false, b3.v, (short)0, acc[0][3], false, false);
        acc[1][0] = __builtin_amdgcn_wmma_f32_16x16x32_bf16(false, a1.v, false, b0.v, (short)0, acc[1][0], false, false);
        acc[1][1] = __builtin_amdgcn_wmma_f32_16x16x32_bf16(false, a1.v, false, b1.v, (short)0, acc[1][1], false, false);
        acc[1][2] = __builtin_amdgcn_wmma_f32_16x16x32_bf16(false, a1.v, false, b2.v, (short)0, acc[1][2], false, false);
        acc[1][3] = __builtin_amdgcn_wmma_f32_16x16x32_bf16(false, a1.v, false, b3.v, (short)0, acc[1][3], false, false);
    }

#pragma unroll
    for (int j = 0; j < 4; ++j) {
        const int colg = n0 + j * 16 + l;
        const float bb = bias ? bias[colg] : 0.0f;
#pragma unroll
        for (int i = 0; i < 2; ++i) {
#pragma unroll
            for (int r = 0; r < 8; ++r) {
                const int rowg = row0 + i * 16 + r + 8 * half;
                epilogue_store(mode, rowg, colg, acc[i][j][r] + bb, N, outF, outH, resid);
            }
        }
    }
}

// ---------------------------------------------------------------------------
// Small 16x16-tile WMMA GEMM (pooler: M == 16).
// ---------------------------------------------------------------------------
__global__ __launch_bounds__(256) void gemm16_wmma_bf16(
    const unsigned short* __restrict__ A, const unsigned short* __restrict__ Bt,
    const float* __restrict__ bias, float* __restrict__ outF,
    unsigned short* __restrict__ outH, const float* __restrict__ resid,
    int M, int N, int K, int mode)
{
    const int wave = threadIdx.x >> 5;
    const int lane = threadIdx.x & 31;
    const int half = lane >> 4;
    const int l    = lane & 15;
    const int tilesN = N >> 4;
    const int tiles  = (M >> 4) * tilesN;
    const int id = blockIdx.x * 8 + wave;
    if (id >= tiles) return;
    const int tm = id / tilesN, tn = id - tm * tilesN;
    const int row0 = tm << 4, n0 = tn << 4;

    const unsigned short* arow = A  + (size_t)(row0 + l) * K + half * 8;
    const unsigned short* brow = Bt + (size_t)(n0  + l) * K + half * 8;

    v8f acc = {};
    for (int k0 = 0; k0 < K; k0 += 32) {
        Frag16 a, b;
        const uint4* ap = (const uint4*)(arow + k0);
        const uint4* bp = (const uint4*)(brow + k0);
        a.u[0] = ap[0]; a.u[1] = ap[2];
        b.u[0] = bp[0]; b.u[1] = bp[2];
        acc = __builtin_amdgcn_wmma_f32_16x16x32_bf16(
                  false, a.v, false, b.v, (short)0, acc, false, false);
    }
    const int colg = n0 + l;
    const float bb = bias ? bias[colg] : 0.0f;
#pragma unroll
    for (int r = 0; r < 8; ++r)
        epilogue_store(mode, row0 + r + 8 * half, colg, acc[r] + bb, N, outF, outH, resid);
}

// ---------------------------------------------------------------------------
// Flash-style attention: one wave per (b, h, 16-row q-tile).
// Online softmax over 32-key chunks; QK^T and P*V via bf16 WMMA;
// probability tile transposed C-layout -> A-layout through LDS.
// ---------------------------------------------------------------------------
__global__ __launch_bounds__(32) void attn_kernel(
    const unsigned short* __restrict__ Q,   // bf16 [B,H,S,64]
    const unsigned short* __restrict__ Kt,  // bf16 [B,H,S,64]
    const unsigned short* __restrict__ Vt,  // bf16 [B,H,64,S]
    const int* __restrict__ input_ids,
    unsigned short* __restrict__ ctx)       // bf16 [B*S, H*64] row-major
{
    const int qt   = blockIdx.x & 31;              // S/16 = 32 q tiles
    const int h    = (blockIdx.x >> 5) % H_DIM;
    const int b    = blockIdx.x / (32 * H_DIM);
    const int lane = threadIdx.x & 31;
    const int half = lane >> 4;
    const int l    = lane & 15;

    const unsigned short* qrow  = Q  + ((((size_t)b * H_DIM + h) * S_DIM + qt * 16 + l) << 6);
    const unsigned short* kbase = Kt + (((size_t)b * H_DIM + h) * S_DIM << 6);
    const unsigned short* vbase = Vt + (((size_t)b * H_DIM + h) << 6) * S_DIM;
    const int* ids = input_ids + (size_t)b * S_DIM;

    Frag16 qa[2];
#pragma unroll
    for (int dc = 0; dc < 2; ++dc) {
        const uint4* p = (const uint4*)(qrow + dc * 32 + half * 8);
        qa[dc].u[0] = p[0]; qa[dc].u[1] = p[2];
    }

    v8f o0 = {}, o1 = {}, o2 = {}, o3 = {};
    float mrow[8], lrow[8];
#pragma unroll
    for (int r = 0; r < 8; ++r) { mrow[r] = -3.0e38f; lrow[r] = 0.0f; }

    __shared__ float pl[16 * 32];

    for (int c = 0; c < S_DIM / 32; ++c) {
        v8f s0 = {}, s1 = {};
#pragma unroll
        for (int dc = 0; dc < 2; ++dc) {
            Frag16 k0f, k1f;
            const uint4* p0 = (const uint4*)(kbase + ((size_t)(c * 32 + l)      << 6) + dc * 32 + half * 8);
            const uint4* p1 = (const uint4*)(kbase + ((size_t)(c * 32 + 16 + l) << 6) + dc * 32 + half * 8);
            k0f.u[0] = p0[0]; k0f.u[1] = p0[2];
            k1f.u[0] = p1[0]; k1f.u[1] = p1[2];
            s0 = __builtin_amdgcn_wmma_f32_16x16x32_bf16(false, qa[dc].v, false, k0f.v, (short)0, s0, false, false);
            s1 = __builtin_amdgcn_wmma_f32_16x16x32_bf16(false, qa[dc].v, false, k1f.v, (short)0, s1, false, false);
        }
        const bool pad0 = (ids[c * 32 + l]      == 0);
        const bool pad1 = (ids[c * 32 + 16 + l] == 0);
        float f0[8], f1[8], alpha[8];
#pragma unroll
        for (int r = 0; r < 8; ++r) {
            f0[r] = pad0 ? -1e9f : s0[r] * 0.125f;   // 1/sqrt(64)
            f1[r] = pad1 ? -1e9f : s1[r] * 0.125f;
        }
#pragma unroll
        for (int r = 0; r < 8; ++r) {
            float cm = fmaxf(f0[r], f1[r]);
            for (int off = 8; off >= 1; off >>= 1)
                cm = fmaxf(cm, __shfl_xor(cm, off, 16));        // 16-lane row group
            float mn = fmaxf(mrow[r], cm);
            alpha[r] = __expf(mrow[r] - mn);
            f0[r] = __expf(f0[r] - mn);
            f1[r] = __expf(f1[r] - mn);
            float rs = f0[r] + f1[r];
            for (int off = 8; off >= 1; off >>= 1)
                rs += __shfl_xor(rs, off, 16);
            lrow[r] = lrow[r] * alpha[r] + rs;
            mrow[r] = mn;
        }
#pragma unroll
        for (int r = 0; r < 8; ++r) {
            o0[r] *= alpha[r]; o1[r] *= alpha[r];
            o2[r] *= alpha[r]; o3[r] *= alpha[r];
        }
        // C-layout probs -> LDS -> A-layout bf16
        __syncthreads();
#pragma unroll
        for (int r = 0; r < 8; ++r) {
            pl[(r + 8 * half) * 32 + l]      = f0[r];
            pl[(r + 8 * half) * 32 + 16 + l] = f1[r];
        }
        __syncthreads();
        Frag16 pa;
#pragma unroll
        for (int i = 0; i < 8; ++i) {
            pa.us[i]     = f2bf(pl[l * 32 + half * 8 + i]);
            pa.us[8 + i] = f2bf(pl[l * 32 + 16 + half * 8 + i]);
        }
        // P(16x32) x V(32x64): 4 output d-tiles
        Frag16 bv;
        const uint4* pv;
        pv = (const uint4*)(vbase + (size_t)(0 * 16 + l) * S_DIM + c * 32 + half * 8);
        bv.u[0] = pv[0]; bv.u[1] = pv[2];
        o0 = __builtin_amdgcn_wmma_f32_16x16x32_bf16(false, pa.v, false, bv.v, (short)0, o0, false, false);
        pv = (const uint4*)(vbase + (size_t)(1 * 16 + l) * S_DIM + c * 32 + half * 8);
        bv.u[0] = pv[0]; bv.u[1] = pv[2];
        o1 = __builtin_amdgcn_wmma_f32_16x16x32_bf16(false, pa.v, false, bv.v, (short)0, o1, false, false);
        pv = (const uint4*)(vbase + (size_t)(2 * 16 + l) * S_DIM + c * 32 + half * 8);
        bv.u[0] = pv[0]; bv.u[1] = pv[2];
        o2 = __builtin_amdgcn_wmma_f32_16x16x32_bf16(false, pa.v, false, bv.v, (short)0, o2, false, false);
        pv = (const uint4*)(vbase + (size_t)(3 * 16 + l) * S_DIM + c * 32 + half * 8);
        bv.u[0] = pv[0]; bv.u[1] = pv[2];
        o3 = __builtin_amdgcn_wmma_f32_16x16x32_bf16(false, pa.v, false, bv.v, (short)0, o3, false, false);
    }

    // normalize and write ctx bf16 [B*S, 768]
#pragma unroll
    for (int r = 0; r < 8; ++r) {
        const float inv = 1.0f / lrow[r];
        const size_t row = (size_t)b * S_DIM + qt * 16 + r + 8 * half;
        unsigned short* dst = ctx + row * (size_t)D_DIM + h * 64 + l;
        dst[0]  = f2bf(o0[r] * inv);
        dst[16] = f2bf(o1[r] * inv);
        dst[32] = f2bf(o2[r] * inv);
        dst[48] = f2bf(o3[r] * inv);
    }
}

// ---------------------------------------------------------------------------
// LayerNorm over rows of y[M,D]; optional fp32 and bf16 outputs.
// ---------------------------------------------------------------------------
__global__ __launch_bounds__(256) void ln_kernel(
    const float* __restrict__ y, const float* __restrict__ g, const float* __restrict__ bia,
    float* __restrict__ outF, unsigned short* __restrict__ outH, int D)
{
    const int row = blockIdx.x, t = threadIdx.x;
    const float* yr = y + (size_t)row * D;
    __shared__ float red[256];
    float s = 0.0f;
    for (int d = t; d < D; d += 256) s += yr[d];
    red[t] = s; __syncthreads();
    for (int o = 128; o > 0; o >>= 1) { if (t < o) red[t] += red[t + o]; __syncthreads(); }
    const float mean = red[0] / D; __syncthreads();
    float v = 0.0f;
    for (int d = t; d < D; d += 256) { float z = yr[d] - mean; v += z * z; }
    red[t] = v; __syncthreads();
    for (int o = 128; o > 0; o >>= 1) { if (t < o) red[t] += red[t + o]; __syncthreads(); }
    const float rstd = rsqrtf(red[0] / D + 1e-5f);
    for (int d = t; d < D; d += 256) {
        float val = (yr[d] - mean) * rstd * g[d] + bia[d];
        if (outF) outF[(size_t)row * D + d] = val;
        if (outH) outH[(size_t)row * D + d] = f2bf(val);
    }
}

// ---------------------------------------------------------------------------
// Embedding gather + LayerNorm (one block per token row).
// ---------------------------------------------------------------------------
__global__ __launch_bounds__(256) void embed_ln_kernel(
    const int* __restrict__ ids, const int* __restrict__ segs,
    const float* __restrict__ tok, const float* __restrict__ pos, const float* __restrict__ seg,
    const float* __restrict__ g, const float* __restrict__ bia,
    float* __restrict__ outF, unsigned short* __restrict__ outH)
{
    const int row = blockIdx.x, t = threadIdx.x;
    const int s = row & (S_DIM - 1);
    const int id = ids[row], sg = segs[row];
    __shared__ float buf[D_DIM];
    __shared__ float red[256];
    float loc = 0.0f;
    for (int d = t; d < D_DIM; d += 256) {
        float e = tok[(size_t)id * D_DIM + d] + pos[(size_t)s * D_DIM + d]
                + seg[(size_t)sg * D_DIM + d];
        buf[d] = e; loc += e;
    }
    red[t] = loc; __syncthreads();
    for (int o = 128; o > 0; o >>= 1) { if (t < o) red[t] += red[t + o]; __syncthreads(); }
    const float mean = red[0] * (1.0f / D_DIM); __syncthreads();
    float v = 0.0f;
    for (int d = t; d < D_DIM; d += 256) { float z = buf[d] - mean; v += z * z; }
    red[t] = v; __syncthreads();
    for (int o = 128; o > 0; o >>= 1) { if (t < o) red[t] += red[t + o]; __syncthreads(); }
    const float rstd = rsqrtf(red[0] * (1.0f / D_DIM) + 1e-5f);
    for (int d = t; d < D_DIM; d += 256) {
        float val = (buf[d] - mean) * rstd * g[d] + bia[d];
        outF[(size_t)row * D_DIM + d] = val;
        outH[(size_t)row * D_DIM + d] = f2bf(val);
    }
}

// ---- prep: f32 -> bf16 convert / transpose-convert ----
__global__ void conv_kernel(const float* __restrict__ src, unsigned short* __restrict__ dst, size_t tot) {
    size_t i = (size_t)blockIdx.x * 256 + threadIdx.x;
    if (i < tot) dst[i] = f2bf(src[i]);
}
__global__ void conv_t_kernel(const float* __restrict__ src, unsigned short* __restrict__ dst, int K, int N) {
    size_t i = (size_t)blockIdx.x * 256 + threadIdx.x;
    size_t tot = (size_t)K * N;
    if (i < tot) { size_t n = i / K, k = i - n * K; dst[i] = f2bf(src[k * (size_t)N + n]); }
}

// ---- gathers ----
__global__ void gather_cls(const float* __restrict__ x, unsigned short* __restrict__ out) {
    const int b = blockIdx.x;
    const float* src = x + (size_t)b * S_DIM * D_DIM;   // s = 0
    for (int d = threadIdx.x; d < D_DIM; d += 256) out[(size_t)b * D_DIM + d] = f2bf(src[d]);
}
__global__ void gather_masked(const float* __restrict__ x, const int* __restrict__ mpos,
                              unsigned short* __restrict__ out) {
    const int i = blockIdx.x;           // 0..B*P-1
    const int b = i / P_DIM, p = i - b * P_DIM;
    const int s = mpos[b * P_DIM + p];
    const float* src = x + ((size_t)b * S_DIM + s) * D_DIM;
    for (int d = threadIdx.x; d < D_DIM; d += 256) out[(size_t)i * D_DIM + d] = f2bf(src[d]);
}

// ---- NSP classifier (tiny) ----
__global__ void nsp_kernel(const unsigned short* __restrict__ pooled,
                           const float* __restrict__ clsW, const float* __restrict__ clsb,
                           float* __restrict__ out) {
    const int t = threadIdx.x;
    if (t >= 32) return;
    const int b = t >> 1, j = t & 1;
    float acc = clsb[j];
    for (int k = 0; k < D_DIM; ++k)
        acc += bf2f(pooled[(size_t)b * D_DIM + k]) * clsW[(size_t)k * 2 + j];
    out[b * 2 + j] = acc;
}

// ---------------------------------------------------------------------------
extern "C" void kernel_launch(void* const* d_in, const int* in_sizes, int n_in,
                              void* d_out, int out_size, void* d_ws, size_t ws_size,
                              hipStream_t stream)
{
    (void)in_sizes; (void)n_in; (void)out_size; (void)ws_size;
    const int*   input_ids   = (const int*)  d_in[0];
    const int*   segment_ids = (const int*)  d_in[1];
    const int*   masked_pos  = (const int*)  d_in[2];
    const float* tok_embed   = (const float*)d_in[3];
    const float* pos_embed   = (const float*)d_in[4];
    const float* seg_embed   = (const float*)d_in[5];
    const float* emb_g       = (const float*)d_in[6];
    const float* emb_b       = (const float*)d_in[7];
    const float* Wq = (const float*)d_in[8];  const float* bq = (const float*)d_in[9];
    const float* Wk = (const float*)d_in[10]; const float* bk = (const float*)d_in[11];
    const float* Wv = (const float*)d_in[12]; const float* bv = (const float*)d_in[13];
    const float* Wo = (const float*)d_in[14]; const float* bo = (const float*)d_in[15];
    const float* ln_g = (const float*)d_in[16]; const float* ln_b = (const float*)d_in[17];
    const float* fc_W = (const float*)d_in[18]; const float* fc_b = (const float*)d_in[19];
    const float* cls_W = (const float*)d_in[20]; const float* cls_b = (const float*)d_in[21];
    const float* mlm_W = (const float*)d_in[22]; const float* mlm_b = (const float*)d_in[23];
    const float* mlm_ln_g = (const float*)d_in[24]; const float* mlm_ln_b = (const float*)d_in[25];
    const float* dec_bias = (const float*)d_in[26];

    // outputs: x [B,S,D] fp32 | logits_lm [B,P,V] | logits_nsp [B,2]
    float* x          = (float*)d_out;
    float* logits_lm  = x + (size_t)NROWS * D_DIM;
    float* logits_nsp = logits_lm + (size_t)B_ * P_DIM * V_DIM;

    // ---- carve workspace ----
    uintptr_t wp = (uintptr_t)d_ws;
    auto alloc = [&](size_t bytes) -> uintptr_t {
        uintptr_t p = wp; wp += (bytes + 255) & ~(size_t)255; return p;
    };
    const size_t WELEM = (size_t)D_DIM * D_DIM;   // 589824
    unsigned short* wtq  = (unsigned short*)alloc(L_DIM * WELEM * 2);
    unsigned short* wtk  = (unsigned short*)alloc(L_DIM * WELEM * 2);
    unsigned short* wtv  = (unsigned short*)alloc(L_DIM * WELEM * 2);
    unsigned short* wto  = (unsigned short*)alloc(L_DIM * WELEM * 2);
    unsigned short* wtfc = (unsigned short*)alloc(WELEM * 2);
    unsigned short* wtml = (unsigned short*)alloc(WELEM * 2);
    unsigned short* tokb = (unsigned short*)alloc((size_t)V_DIM * D_DIM * 2);
    unsigned short* xb   = (unsigned short*)alloc((size_t)NROWS * D_DIM * 2);
    unsigned short* qb   = (unsigned short*)alloc((size_t)NROWS * D_DIM * 2);
    unsigned short* kb   = (unsigned short*)alloc((size_t)NROWS * D_DIM * 2);
    unsigned short* vtb  = (unsigned short*)alloc((size_t)NROWS * D_DIM * 2);
    unsigned short* ctxb = (unsigned short*)alloc((size_t)NROWS * D_DIM * 2);
    unsigned short* poolin = (unsigned short*)alloc((size_t)B_ * D_DIM * 2);
    unsigned short* pooled = (unsigned short*)alloc((size_t)B_ * D_DIM * 2);
    unsigned short* hm_in  = (unsigned short*)alloc((size_t)B_ * P_DIM * D_DIM * 2);
    float*          hm_gel = (float*)alloc((size_t)B_ * P_DIM * D_DIM * 4);
    unsigned short* hm_ln  = (unsigned short*)alloc((size_t)B_ * P_DIM * D_DIM * 2);
    // pre-LN fp32 y overlays the (dead after attention) qb+kb region: 25.2 MB
    float* yf = (float*)qb;

    // ---- prep: weight transpose-converts + tok_embed convert ----
    const int tgrid = (int)((WELEM + 255) / 256);
    for (int i = 0; i < L_DIM; ++i) {
        conv_t_kernel<<<tgrid, 256, 0, stream>>>(Wq + i * WELEM, wtq + i * WELEM, D_DIM, D_DIM);
        conv_t_kernel<<<tgrid, 256, 0, stream>>>(Wk + i * WELEM, wtk + i * WELEM, D_DIM, D_DIM);
        conv_t_kernel<<<tgrid, 256, 0, stream>>>(Wv + i * WELEM, wtv + i * WELEM, D_DIM, D_DIM);
        conv_t_kernel<<<tgrid, 256, 0, stream>>>(Wo + i * WELEM, wto + i * WELEM, D_DIM, D_DIM);
    }
    conv_t_kernel<<<tgrid, 256, 0, stream>>>(fc_W,  wtfc, D_DIM, D_DIM);
    conv_t_kernel<<<tgrid, 256, 0, stream>>>(mlm_W, wtml, D_DIM, D_DIM);
    const size_t tokElems = (size_t)V_DIM * D_DIM;
    conv_kernel<<<(int)((tokElems + 255) / 256), 256, 0, stream>>>(tok_embed, tokb, tokElems);

    // ---- embeddings + LN ----
    embed_ln_kernel<<<NROWS, 256, 0, stream>>>(input_ids, segment_ids, tok_embed,
                                               pos_embed, seg_embed, emb_g, emb_b, x, xb);

    // ---- encoder layers ----
    // 32x64 tiles: (8192/32)*(768/64) = 3072 waves -> 384 blocks of 8 waves
    const int gblocks = ((NROWS / 32) * (D_DIM / 64) + 7) / 8;
    for (int i = 0; i < L_DIM; ++i) {
        gemm32x64_wmma_bf16<<<gblocks, 256, 0, stream>>>(xb, wtq + i * WELEM, bq + i * D_DIM,
                                                         nullptr, qb, nullptr, NROWS, D_DIM, D_DIM, 1);
        gemm32x64_wmma_bf16<<<gblocks, 256, 0, stream>>>(xb, wtk + i * WELEM, bk + i * D_DIM,
                                                         nullptr, kb, nullptr, NROWS, D_DIM, D_DIM, 1);
        gemm32x64_wmma_bf16<<<gblocks, 256, 0, stream>>>(xb, wtv + i * WELEM, bv + i * D_DIM,
                                                         nullptr, vtb, nullptr, NROWS, D_DIM, D_DIM, 2);
        attn_kernel<<<B_ * H_DIM * (S_DIM / 16), 32, 0, stream>>>(qb, kb, vtb, input_ids, ctxb);
        gemm32x64_wmma_bf16<<<gblocks, 256, 0, stream>>>(ctxb, wto + i * WELEM, bo + i * D_DIM,
                                                         yf, nullptr, x, NROWS, D_DIM, D_DIM, 3);
        ln_kernel<<<NROWS, 256, 0, stream>>>(yf, ln_g + i * D_DIM, ln_b + i * D_DIM, x, xb, D_DIM);
    }

    // ---- NSP head ----
    gather_cls<<<B_, 256, 0, stream>>>(x, poolin);
    gemm16_wmma_bf16<<<6, 256, 0, stream>>>(poolin, wtfc, fc_b, nullptr, pooled, nullptr,
                                            16, D_DIM, D_DIM, 5);
    nsp_kernel<<<1, 32, 0, stream>>>(pooled, cls_W, cls_b, logits_nsp);

    // ---- MLM head ----
    gather_masked<<<B_ * P_DIM, 256, 0, stream>>>(x, masked_pos, hm_in);
    const int mblocks = (((B_ * P_DIM) / 32) * (D_DIM / 64) + 7) / 8;     // 15
    gemm32x64_wmma_bf16<<<mblocks, 256, 0, stream>>>(hm_in, wtml, mlm_b, hm_gel, nullptr, nullptr,
                                                     (B_ * P_DIM), D_DIM, D_DIM, 6);
    ln_kernel<<<B_ * P_DIM, 256, 0, stream>>>(hm_gel, mlm_ln_g, mlm_ln_b, nullptr, hm_ln, D_DIM);
    const int dblocks = (((B_ * P_DIM) / 32) * (V_DIM / 64) + 7) / 8;     // 625
    gemm32x64_wmma_bf16<<<dblocks, 256, 0, stream>>>(hm_ln, tokb, dec_bias, logits_lm, nullptr, nullptr,
                                                     (B_ * P_DIM), V_DIM, D_DIM, 0);
}
